// CrossAttentionFusion_27178553049173
// MI455X (gfx1250) — matrix operand order repaired and checked by reading, not compiled, not run on hardware
//
#include <hip/hip_runtime.h>

// CrossAttentionFusion for MI455X (gfx1250, wave32, WMMA).
// All GEMMs: M=4096, N=1024, K=1024 in f16 with f32 accumulation via
// v_wmma_f32_16x16x32_f16. Flash-attention keeps the 256MB score tensor
// entirely on-chip. ~52 GFLOP total vs ~100MB HBM traffic -> memory bound;
// f16 intermediates + fused epilogues minimize bytes moved.

typedef __attribute__((ext_vector_type(16))) _Float16 v16h;
typedef __attribute__((ext_vector_type(8)))  float    v8f;

#define D_MODEL  1024
#define N_HEADS  16
#define HEAD_DIM 64
#define B_SZ     4
#define NQ       1024
#define NKV      1024
#define M_TOT    (B_SZ * NQ)   /* 4096 */

// ---------------------------------------------------------------------------
// WMMA fragment loaders (wave32 register layouts per CDNA5 ISA 7.12.2).
// A 16x32 f16: lane L (l=L&15, h=L>>4): row = mbase+l
//   elems 0..7  -> k = koff + 8*h  .. +7
//   elems 8..15 -> k = koff + 16+8*h .. +7
// B 32x16 f16 staged as BT rows (n-major, k-contiguous): lane L: n = nbase+l,
//   elems 0..7 -> k = koff + 8*h .. +7, elems 8..15 -> k = koff + 16+8*h .. +7
// Strides are padded (multiples of 8 halves) so every read is a ds_load_b128.
// ---------------------------------------------------------------------------
__device__ __forceinline__ v16h frag_a(const _Float16* base, int stride, int mbase, int koff) {
  const int lane = threadIdx.x & 31;
  const _Float16* p = base + (mbase + (lane & 15)) * stride + koff + ((lane >> 4) << 3);
  union { v16h v; float4 f[2]; } u;
  u.f[0] = *(const float4*)(p);
  u.f[1] = *(const float4*)(p + 16);
  return u.v;
}

__device__ __forceinline__ v16h frag_b(const _Float16* base, int stride, int nbase, int koff) {
  const int lane = threadIdx.x & 31;
  const _Float16* p = base + (nbase + (lane & 15)) * stride + koff + ((lane >> 4) << 3);
  union { v16h v; float4 f[2]; } u;
  u.f[0] = *(const float4*)(p);
  u.f[1] = *(const float4*)(p + 16);
  return u.v;
}

__device__ __forceinline__ v8f wmma16(v16h a, v16h b, v8f c) {
  return __builtin_amdgcn_wmma_f32_16x16x32_f16(false, a, false, b, (short)0, c, false, false);
}

// ---------------------------------------------------------------------------
// f32 -> f16 elementwise convert (activations)
// ---------------------------------------------------------------------------
__global__ void __launch_bounds__(256) cvt_f32_f16(const float* __restrict__ src,
                                                   _Float16* __restrict__ dst, int n) {
  const int i = (blockIdx.x * 256 + threadIdx.x) * 4;
  if (i + 3 < n) {
    float4 v = *(const float4*)(src + i);
    dst[i + 0] = (_Float16)v.x;
    dst[i + 1] = (_Float16)v.y;
    dst[i + 2] = (_Float16)v.z;
    dst[i + 3] = (_Float16)v.w;
  }
}

// ---------------------------------------------------------------------------
// Transposing f32 -> f16 weight convert: WT[n][k] = (f16)W[k][n], D=1024.
// Classic 32x32 LDS tile so both global sides are coalesced.
// ---------------------------------------------------------------------------
__global__ void __launch_bounds__(256) transpose_cvt(const float* __restrict__ src,
                                                     _Float16* __restrict__ dst) {
  __shared__ float tile[32][33];
  const int x  = blockIdx.x * 32 + threadIdx.x;   // src column
  const int y0 = blockIdx.y * 32;                 // src row base
  for (int i = threadIdx.y; i < 32; i += 8)
    tile[i][threadIdx.x] = src[(size_t)(y0 + i) * D_MODEL + x];
  __syncthreads();
  const int ox  = blockIdx.y * 32 + threadIdx.x;  // dst column (= src row)
  const int oy0 = blockIdx.x * 32;                // dst row base (= src col)
  for (int i = threadIdx.y; i < 32; i += 8)
    dst[(size_t)(oy0 + i) * D_MODEL + ox] = (_Float16)tile[threadIdx.x][i];
}

// ---------------------------------------------------------------------------
// WMMA GEMM: C[4096x1024] = A[4096x1024] * BT^T + bias, K-tiled by 32.
// Block = 256 threads (8 waves), tile 128x128, each wave does 32x64 (8 accums).
// Double-buffered LDS. MODE controls the epilogue:
//   0: f16 -> out[((b*16+h)*1024+m)*64+d]            (Q, K: [B][H][N][64])
//   1: f16 -> out[((b*16+h)*64+d)*1024+m]            (V transposed: [B][H][64][Nkv])
//   2: f32 -> out[m*1024+n] = acc + bias + resid     (O-proj + residual)
// ---------------------------------------------------------------------------
#define KP 40   // padded LDS k-stride in halves (80B rows: 16B aligned, bank-spread)

template<int MODE>
__global__ void __launch_bounds__(256) gemm_kernel(const _Float16* __restrict__ A,
                                                   const _Float16* __restrict__ BT,
                                                   const float* __restrict__ bias,
                                                   const float* __restrict__ resid,
                                                   void* __restrict__ out) {
  __shared__ _Float16 As[2][128 * KP];
  __shared__ _Float16 Bs[2][128 * KP];

  const int m0   = blockIdx.y * 128;
  const int n0   = blockIdx.x * 128;
  const int t    = threadIdx.x;
  const int lane = t & 31;
  const int wave = t >> 5;
  const int wm   = (wave >> 1) * 32;   // wave row base within tile
  const int wn   = (wave & 1) * 64;    // wave col base within tile

  v8f acc[2][4];
  const v8f vzero = {0.f, 0.f, 0.f, 0.f, 0.f, 0.f, 0.f, 0.f};
#pragma unroll
  for (int i = 0; i < 2; ++i)
#pragma unroll
    for (int j = 0; j < 4; ++j) acc[i][j] = vzero;

  auto load_tile = [&](int kt, int buf) {
    const int r = t >> 1;            // 0..127
    const int c = (t & 1) * 16;      // 0 or 16 (halves)
    const float4* ga = (const float4*)(A + (size_t)(m0 + r) * D_MODEL + kt * 32 + c);
    float4* la = (float4*)(As[buf] + r * KP + c);
    la[0] = ga[0]; la[1] = ga[1];
    const float4* gb = (const float4*)(BT + (size_t)(n0 + r) * D_MODEL + kt * 32 + c);
    float4* lb = (float4*)(Bs[buf] + r * KP + c);
    lb[0] = gb[0]; lb[1] = gb[1];
    if (kt + 2 < 32) {  // L2 prefetch two tiles ahead -> global_prefetch_b8
      __builtin_prefetch(A  + (size_t)(m0 + r) * D_MODEL + (kt + 2) * 32 + c, 0, 1);
      __builtin_prefetch(BT + (size_t)(n0 + r) * D_MODEL + (kt + 2) * 32 + c, 0, 1);
    }
  };

  load_tile(0, 0);
  int buf = 0;
  for (int kt = 0; kt < 32; ++kt) {
    __syncthreads();
    if (kt + 1 < 32) load_tile(kt + 1, buf ^ 1);

    const v16h a0 = frag_a(As[buf], KP, wm, 0);
    const v16h a1 = frag_a(As[buf], KP, wm + 16, 0);
#pragma unroll
    for (int j = 0; j < 4; ++j) {
      const v16h b = frag_b(Bs[buf], KP, wn + j * 16, 0);
      acc[0][j] = wmma16(a0, b, acc[0][j]);
      acc[1][j] = wmma16(a1, b, acc[1][j]);
    }
    buf ^= 1;
  }

  // Epilogue. C layout: elem v of lane L -> m = v + 8*(L>>4), n = L&15.
#pragma unroll
  for (int i = 0; i < 2; ++i)
#pragma unroll
    for (int j = 0; j < 4; ++j) {
      const int gn = n0 + wn + j * 16 + (lane & 15);
      const float bv = bias[gn];
#pragma unroll
      for (int v = 0; v < 8; ++v) {
        const int gm = m0 + wm + i * 16 + v + ((lane >> 4) << 3);
        const float val = acc[i][j][v] + bv;
        if (MODE == 0) {
          const int b = gm >> 10, mq = gm & 1023;
          const int h = gn >> 6,  d  = gn & 63;
          ((_Float16*)out)[((size_t)((b * N_HEADS + h) << 10) + mq) * HEAD_DIM + d] = (_Float16)val;
        } else if (MODE == 1) {
          const int b = gm >> 10, mkv = gm & 1023;
          const int h = gn >> 6,  d   = gn & 63;
          ((_Float16*)out)[((size_t)(b * N_HEADS + h) * HEAD_DIM + d) * NKV + mkv] = (_Float16)val;
        } else {
          ((float*)out)[(size_t)gm * D_MODEL + gn] = val + resid[(size_t)gm * D_MODEL + gn];
        }
      }
    }
}

// ---------------------------------------------------------------------------
// Flash attention, non-causal. Grid (Nq/128, H, B), block 256 (8 waves).
// Each wave owns 16 q rows; kv tiled by 64. Scores/PV via WMMA f16;
// online softmax in f32 with shfl_xor row reductions (16-lane groups match
// the C-matrix layout). P converts C-layout -> A-fragment layout via LDS.
// ---------------------------------------------------------------------------
#define SP 72   // padded LDS stride (halves) for 64-wide tiles

__global__ void __launch_bounds__(256) attn_kernel(const _Float16* __restrict__ Q16,
                                                   const _Float16* __restrict__ K16,
                                                   const _Float16* __restrict__ VT16,
                                                   _Float16* __restrict__ AO16) {
  __shared__ _Float16 Qs[128 * SP];
  __shared__ _Float16 Ks[64 * SP];
  __shared__ _Float16 Vs[64 * SP];   // V^T tile: rows = d, cols = kv
  __shared__ _Float16 Ps[128 * SP];  // wave w owns rows 16w..16w+15

  const int qb   = blockIdx.x;            // q block (128 rows)
  const int h    = blockIdx.y;
  const int b    = blockIdx.z;
  const int t    = threadIdx.x;
  const int lane = t & 31;
  const int wave = t >> 5;
  const int lhi  = lane >> 4;             // half-wave id

  const _Float16* qp = Q16  + ((size_t)(b * N_HEADS + h) * NQ + qb * 128) * HEAD_DIM;
  const _Float16* kp = K16  +  (size_t)(b * N_HEADS + h) * NKV * HEAD_DIM;
  const _Float16* vp = VT16 +  (size_t)(b * N_HEADS + h) * HEAD_DIM * NKV;

  // Stage the 128x64 Q block once.
  for (int idx = t; idx < 1024; idx += 256) {
    const int r = idx >> 3, c = (idx & 7) * 8;
    *(float4*)(Qs + r * SP + c) = *(const float4*)(qp + r * HEAD_DIM + c);
  }
  __syncthreads();

  // Wave-resident Q fragments (k = head dim, two 32-wide windows).
  const v16h qa0 = frag_a(Qs, SP, wave * 16, 0);
  const v16h qa1 = frag_a(Qs, SP, wave * 16, 32);

  const v8f vzero = {0.f, 0.f, 0.f, 0.f, 0.f, 0.f, 0.f, 0.f};
  v8f o[4];
  float mrun[8], lrun[8];
#pragma unroll
  for (int j = 0; j < 4; ++j) o[j] = vzero;
#pragma unroll
  for (int v = 0; v < 8; ++v) { mrun[v] = -3.0e38f; lrun[v] = 0.f; }

  for (int kt = 0; kt < NKV / 64; ++kt) {
    __syncthreads();  // prior iteration's WMMA reads of Ks/Vs/Ps are done
    for (int idx = t; idx < 512; idx += 256) {
      const int r = idx >> 3, c = (idx & 7) * 8;
      *(float4*)(Ks + r * SP + c) = *(const float4*)(kp + (size_t)(kt * 64 + r) * HEAD_DIM + c);
    }
    for (int idx = t; idx < 512; idx += 256) {
      const int r = idx >> 3, c = (idx & 7) * 8;
      *(float4*)(Vs + r * SP + c) = *(const float4*)(vp + (size_t)r * NKV + kt * 64 + c);
    }
    __syncthreads();

    // S (16 q x 64 kv) = Q * K^T : K rows are k-contiguous -> direct B frags.
    v8f s_acc[4];
#pragma unroll
    for (int j = 0; j < 4; ++j) s_acc[j] = vzero;
#pragma unroll
    for (int j = 0; j < 4; ++j) {
      s_acc[j] = wmma16(qa0, frag_b(Ks, SP, j * 16, 0),  s_acc[j]);
      s_acc[j] = wmma16(qa1, frag_b(Ks, SP, j * 16, 32), s_acc[j]);
    }

    // Online softmax per row (row m = v + 8*lhi lives in 16 lanes of one half).
#pragma unroll
    for (int v = 0; v < 8; ++v) {
      float mx = -3.0e38f;
#pragma unroll
      for (int j = 0; j < 4; ++j) {
        s_acc[j][v] *= 0.125f;                 // 1/sqrt(HEAD_DIM)
        mx = fmaxf(mx, s_acc[j][v]);
      }
#pragma unroll
      for (int off = 1; off < 16; off <<= 1) mx = fmaxf(mx, __shfl_xor(mx, off));
      const float mnew  = fmaxf(mrun[v], mx);
      const float alpha = __expf(mrun[v] - mnew);
      float rs = 0.f;
#pragma unroll
      for (int j = 0; j < 4; ++j) {
        const float pe = __expf(s_acc[j][v] - mnew);
        rs += pe;
        Ps[(wave * 16 + v + (lhi << 3)) * SP + j * 16 + (lane & 15)] = (_Float16)pe;
      }
#pragma unroll
      for (int off = 1; off < 16; off <<= 1) rs += __shfl_xor(rs, off);
      lrun[v] = lrun[v] * alpha + rs;
      mrun[v] = mnew;
#pragma unroll
      for (int j = 0; j < 4; ++j) o[j][v] *= alpha;
    }
    __syncthreads();  // P stores visible before A-fragment reads

    // O += P * V : Vs rows (= d) are kv-contiguous -> direct B frags.
    const v16h pa0 = frag_a(Ps, SP, wave * 16, 0);
    const v16h pa1 = frag_a(Ps, SP, wave * 16, 32);
#pragma unroll
    for (int j = 0; j < 4; ++j) {
      o[j] = wmma16(pa0, frag_b(Vs, SP, j * 16, 0),  o[j]);
      o[j] = wmma16(pa1, frag_b(Vs, SP, j * 16, 32), o[j]);
    }
  }

  // Normalize and write AO as [B][Nq][H*64] (row-major D -> ready for Wo GEMM).
#pragma unroll
  for (int j = 0; j < 4; ++j) {
    const int d = j * 16 + (lane & 15);
#pragma unroll
    for (int v = 0; v < 8; ++v) {
      const int q = qb * 128 + wave * 16 + v + (lhi << 3);
      AO16[((size_t)(b * NQ + q) * N_HEADS + h) * HEAD_DIM + d] = (_Float16)(o[j][v] / lrun[v]);
    }
  }
}

// ---------------------------------------------------------------------------
// Row LayerNorm over D=1024: out = (x - mu) * rsqrt(var + 1e-5) * gamma + beta
// ---------------------------------------------------------------------------
__global__ void __launch_bounds__(256) layernorm_kernel(const float* __restrict__ X,
                                                        const float* __restrict__ gamma,
                                                        const float* __restrict__ beta,
                                                        float* __restrict__ out) {
  const int row = blockIdx.x;
  const float* x = X + (size_t)row * D_MODEL;
  float s = 0.f, s2 = 0.f;
  for (int i = threadIdx.x; i < D_MODEL; i += 256) {
    const float v = x[i];
    s += v; s2 += v * v;
  }
#pragma unroll
  for (int off = 16; off; off >>= 1) { s += __shfl_xor(s, off); s2 += __shfl_xor(s2, off); }
  __shared__ float ws1[8], ws2[8];
  const int wave = threadIdx.x >> 5, lane = threadIdx.x & 31;
  if (lane == 0) { ws1[wave] = s; ws2[wave] = s2; }
  __syncthreads();
  if (wave == 0) {
    s  = (lane < 8) ? ws1[lane] : 0.f;
    s2 = (lane < 8) ? ws2[lane] : 0.f;
#pragma unroll
    for (int off = 4; off; off >>= 1) { s += __shfl_xor(s, off); s2 += __shfl_xor(s2, off); }
    if (lane == 0) { ws1[0] = s; ws2[0] = s2; }
  }
  __syncthreads();
  const float mu   = ws1[0] * (1.0f / D_MODEL);
  const float var  = ws2[0] * (1.0f / D_MODEL) - mu * mu;
  const float rstd = rsqrtf(var + 1e-5f);
  for (int i = threadIdx.x; i < D_MODEL; i += 256)
    out[(size_t)row * D_MODEL + i] = (x[i] - mu) * rstd * gamma[i] + beta[i];
}

// ---------------------------------------------------------------------------
// Host-side orchestration (graph-capture safe: only kernel launches on stream)
// ---------------------------------------------------------------------------
extern "C" void kernel_launch(void* const* d_in, const int* in_sizes, int n_in,
                              void* d_out, int out_size, void* d_ws, size_t ws_size,
                              hipStream_t stream) {
  (void)in_sizes; (void)n_in; (void)out_size; (void)ws_size;
  const float* query = (const float*)d_in[0];
  const float* keyv  = (const float*)d_in[1];
  const float* Wq    = (const float*)d_in[2];
  const float* bq    = (const float*)d_in[3];
  const float* Wk    = (const float*)d_in[4];
  const float* bk    = (const float*)d_in[5];
  const float* Wv    = (const float*)d_in[6];
  const float* bv    = (const float*)d_in[7];
  const float* Wo    = (const float*)d_in[8];
  const float* bo    = (const float*)d_in[9];
  const float* gamma = (const float*)d_in[10];
  const float* beta  = (const float*)d_in[11];
  float* out = (float*)d_out;

  // Workspace carve-up (total 72 MiB, all 16B-aligned).
  char* p = (char*)d_ws;
  auto take = [&](size_t bytes) { char* r = p; p += bytes; return r; };
  const size_t ACT = (size_t)M_TOT * D_MODEL;         // 4M elements
  const size_t WEL = (size_t)D_MODEL * D_MODEL;       // 1M elements
  _Float16* qf16  = (_Float16*)take(ACT * 2);
  _Float16* kvf16 = (_Float16*)take(ACT * 2);
  _Float16* WqT   = (_Float16*)take(WEL * 2);
  _Float16* WkT   = (_Float16*)take(WEL * 2);
  _Float16* WvT   = (_Float16*)take(WEL * 2);
  _Float16* WoT   = (_Float16*)take(WEL * 2);
  _Float16* Q16   = (_Float16*)take(ACT * 2);
  _Float16* K16   = (_Float16*)take(ACT * 2);
  _Float16* VT16  = (_Float16*)take(ACT * 2);
  _Float16* AO16  = (_Float16*)take(ACT * 2);
  float*    X     = (float*)take(ACT * 4);

  // 1) precision conversion (+ weight transpose)
  cvt_f32_f16<<<ACT / 1024, 256, 0, stream>>>(query, qf16, (int)ACT);
  cvt_f32_f16<<<ACT / 1024, 256, 0, stream>>>(keyv, kvf16, (int)ACT);
  dim3 tgrid(D_MODEL / 32, D_MODEL / 32);
  dim3 tblk(32, 8);
  transpose_cvt<<<tgrid, tblk, 0, stream>>>(Wq, WqT);
  transpose_cvt<<<tgrid, tblk, 0, stream>>>(Wk, WkT);
  transpose_cvt<<<tgrid, tblk, 0, stream>>>(Wv, WvT);
  transpose_cvt<<<tgrid, tblk, 0, stream>>>(Wo, WoT);

  // 2) Q/K/V projections (WMMA GEMMs, fused bias + head-layout scatter)
  dim3 ggrid(D_MODEL / 128, M_TOT / 128);   // (8, 32)
  gemm_kernel<0><<<ggrid, 256, 0, stream>>>(qf16,  WqT, bq, nullptr, (void*)Q16);
  gemm_kernel<0><<<ggrid, 256, 0, stream>>>(kvf16, WkT, bk, nullptr, (void*)K16);
  gemm_kernel<1><<<ggrid, 256, 0, stream>>>(kvf16, WvT, bv, nullptr, (void*)VT16);

  // 3) flash attention (scores never hit HBM)
  dim3 agrid(NQ / 128, N_HEADS, B_SZ);      // (8, 16, 4)
  attn_kernel<<<agrid, 256, 0, stream>>>(Q16, K16, VT16, AO16);

  // 4) output projection + bias + residual (f32), then LayerNorm -> d_out
  gemm_kernel<2><<<ggrid, 256, 0, stream>>>(AO16, WoT, bo, query, (void*)X);
  layernorm_kernel<<<M_TOT, 256, 0, stream>>>(X, gamma, beta, out);
}
